// Attention_10548439679153
// MI455X (gfx1250) — matrix operand order repaired and checked by reading, not compiled, not run on hardware
//
#include <hip/hip_runtime.h>
#include <cstdint>
#include <cstddef>

// ---------------------------------------------------------------------------
// CDNA5 / gfx1250 attention block, bf16 WMMA pipeline + TDM tile staging.
// ---------------------------------------------------------------------------

typedef __attribute__((ext_vector_type(16))) __bf16 v16bf;
typedef __attribute__((ext_vector_type(8)))  float  v8f;
typedef __attribute__((ext_vector_type(4)))  unsigned int v4u;
typedef __attribute__((ext_vector_type(8)))  int v8i;
typedef __attribute__((ext_vector_type(4)))  int v4i;

__device__ __forceinline__ unsigned short f2bf(float f) {
  union { float f; unsigned u; } x; x.f = f;
  unsigned r = x.u + 0x7FFFu + ((x.u >> 16) & 1u);   // round-to-nearest-even
  return (unsigned short)(r >> 16);
}
__device__ __forceinline__ float bf2f(unsigned short b) {
  union { unsigned u; float f; } x; x.u = ((unsigned)b) << 16; return x.f;
}
__device__ __forceinline__ __bf16 us2bf(unsigned short u) {
  return __builtin_bit_cast(__bf16, u);
}

// ---------------------------------------------------------------------------
// TDM: async DMA of a 2-D bf16 tile [rows x 32] from a row-major tensor with
// row stride `strideElems` into LDS at `lds_off`, padded +16B per 64B row
// (matches As[rows][32+8] ushort layout). Issued once per calling wave.
// ---------------------------------------------------------------------------
__device__ __forceinline__ void tdm_load_tile_A(const unsigned short* gptr,
                                                unsigned lds_off,
                                                int dim0_rem, int dim1_rem,
                                                int strideElems) {
  unsigned long long ga = (unsigned long long)(uintptr_t)gptr;
  v4u g0;
  g0.x = 0x1u;                                            // count=1, user mode
  g0.y = lds_off;                                         // lds_addr (bytes)
  g0.z = (unsigned)(ga & 0xFFFFFFFFu);                    // global_addr[31:0]
  g0.w = (unsigned)((ga >> 32) & 0x01FFFFFFu)             // global_addr[56:32]
         | (2u << 30);                                    // type=2 (image)
  v8i g1;
  // data_size=1 (2B), pad_enable, pad_interval=3 (16 DW), pad_amount=3 (4 DW)
  g1[0] = (int)((1u << 16) | (1u << 20) | (3u << 22) | (3u << 25));
  g1[1] = (int)(((unsigned)dim0_rem & 0xFFFFu) << 16);    // tensor_dim0[15:0]
  g1[2] = (int)((((unsigned)dim0_rem >> 16) & 0xFFFFu)    // tensor_dim0[31:16]
         | (((unsigned)dim1_rem & 0xFFFFu) << 16));       // tensor_dim1[15:0]
  g1[3] = (int)((((unsigned)dim1_rem >> 16) & 0xFFFFu)    // tensor_dim1[31:16]
         | (32u << 16));                                  // tile_dim0 = 32
  g1[4] = (int)128u;                                      // tile_dim1=128, tile_dim2=0
  g1[5] = strideElems;                                    // tensor_dim0_stride[31:0]
  g1[6] = 0;                                              // stride hi / dim1_stride lo
  g1[7] = 0;
  v4i z4 = {0, 0, 0, 0};
  v8i z8 = {0, 0, 0, 0, 0, 0, 0, 0};
  // 6-arg form (clang-23 / therock-10.0 headers)
  __builtin_amdgcn_tensor_load_to_lds(g0, g1, z4, z4, z8, 0);
}

// ---------------------------------------------------------------------------
// fp32 -> bf16 (raw ushort) convert
// ---------------------------------------------------------------------------
__global__ __launch_bounds__(256) void k_f32_to_bf16(const float* __restrict__ in,
                                                     unsigned short* __restrict__ out,
                                                     int n) {
  int i = blockIdx.x * 256 + threadIdx.x;
  if (i < n) out[i] = f2bf(in[i]);
}

// ---------------------------------------------------------------------------
// RoPE on bf16 tensor laid out [rows, nheads*64]; pos = row % 1024.
// freqs_cos/sin: [1024, 32] fp32.
// ---------------------------------------------------------------------------
__global__ __launch_bounds__(256) void k_rope_bf16(unsigned short* __restrict__ t,
                                                   const float* __restrict__ cosT,
                                                   const float* __restrict__ sinT,
                                                   int nheads, int total_pairs) {
  int i = blockIdx.x * 256 + threadIdx.x;
  if (i >= total_pairs) return;
  int ppr = nheads * 32;                 // pairs per row
  int row = i / ppr;
  int c   = i - row * ppr;
  int h   = c >> 5;
  int j   = c & 31;
  int pos = row & 1023;                  // S = 1024
  size_t base = (size_t)row * nheads * 64 + (size_t)h * 64 + 2 * j;
  float t0 = bf2f(t[base]), t1 = bf2f(t[base + 1]);
  float cv = cosT[pos * 32 + j], sv = sinT[pos * 32 + j];
  t[base]     = f2bf(t0 * cv - t1 * sv);
  t[base + 1] = f2bf(t0 * sv + t1 * cv);
}

// ---------------------------------------------------------------------------
// bf16 WMMA GEMM: C[M,N] = A[M,K] * B[K,N], A/B bf16(row-major), C fp32 or bf16.
// Block tile 128x64, 8 waves (wave grid 4x2), wave tile 32x32 (2x2 WMMA tiles),
// K step 32. A tile staged by the Tensor Data Mover (async, padded rows);
// B tile transposed into LDS by the waves while the TDM runs.
// ---------------------------------------------------------------------------
template <bool OUT_F32>
__global__ __launch_bounds__(256)
void k_gemm_bf16(const unsigned short* __restrict__ A,
                 const unsigned short* __restrict__ B,
                 void* __restrict__ C, int M, int N, int K) {
  constexpr int BM = 128, BN = 64, BK = 32;
  __shared__ unsigned short As[BM][BK + 8];   // [m][k]  (TDM destination)
  __shared__ unsigned short Bs[BN][BK + 8];   // [n][k]  (transposed)

  const int tid  = threadIdx.x;
  const int lane = tid & 31;
  const int wave = tid >> 5;
  const int wm   = wave & 3;        // 0..3 -> M
  const int wn   = wave >> 2;       // 0..1 -> N
  const int hlf  = lane >> 4;       // lane half (0/1)
  const int l16  = lane & 15;
  const int bm   = blockIdx.x * BM;
  const int bn   = blockIdx.y * BN;
  const unsigned as_lds = (unsigned)(uintptr_t)&As[0][0];

  v8f acc[2][2];
  const v8f vzero = {0.f, 0.f, 0.f, 0.f, 0.f, 0.f, 0.f, 0.f};
#pragma unroll
  for (int i = 0; i < 2; ++i)
#pragma unroll
    for (int j = 0; j < 2; ++j) acc[i][j] = vzero;

  for (int k0 = 0; k0 < K; k0 += BK) {
    // ---- async A tile (128x32) via Tensor Data Mover, wave 0 issues
    if (wave == 0)
      tdm_load_tile_A(A + (size_t)bm * K + k0, as_lds, K - k0, M - bm, K);

    // ---- stage B tile (32x64) transposed into Bs[n][k] meanwhile
#pragma unroll
    for (int c = tid; c < BK * BN / 8; c += 256) {
      int kk = c >> 3;
      int nn = (c & 7) * 8;
      const unsigned short* gp = B + (size_t)(k0 + kk) * N + bn + nn;
      if (k0 + BK < K) __builtin_prefetch(gp + (size_t)BK * N, 0, 0);
      uint4 v = *(const uint4*)gp;
      const unsigned short* pv = (const unsigned short*)&v;
#pragma unroll
      for (int j = 0; j < 8; ++j) Bs[nn + j][kk] = pv[j];
    }
    if (wave == 0) __builtin_amdgcn_s_wait_tensorcnt(0);
    __syncthreads();

    // ---- build fragments (ISA 7.12.2 layouts) and issue 4 WMMAs
    v16bf af[2], bf[2];
#pragma unroll
    for (int i = 0; i < 2; ++i) {
      const unsigned short* ap = &As[wm * 32 + i * 16 + l16][hlf * 8];
#pragma unroll
      for (int e = 0; e < 8; ++e) {
        af[i][e]     = us2bf(ap[e]);        // K = hlf*8 + e
        af[i][8 + e] = us2bf(ap[16 + e]);   // K = 16 + hlf*8 + e
      }
    }
#pragma unroll
    for (int j = 0; j < 2; ++j) {
      const unsigned short* bp = &Bs[wn * 32 + j * 16 + l16][hlf * 8];
#pragma unroll
      for (int e = 0; e < 16; ++e) bf[j][e] = us2bf(bp[(e & 7) + ((e >> 3) << 4)]);  // K = hlf*8 + e (e<8), 16 + hlf*8 + e-8
    }
#pragma unroll
    for (int i = 0; i < 2; ++i)
#pragma unroll
      for (int j = 0; j < 2; ++j)
        acc[i][j] = __builtin_amdgcn_wmma_f32_16x16x32_bf16(
            false, af[i], false, bf[j], (short)0, acc[i][j], false, false);
    __syncthreads();
  }

  // epilogue: C layout -> row = v + 8*hlf, col = l16 within each 16x16 tile
#pragma unroll
  for (int i = 0; i < 2; ++i)
#pragma unroll
    for (int j = 0; j < 2; ++j)
#pragma unroll
      for (int v = 0; v < 8; ++v) {
        int row = bm + wm * 32 + i * 16 + hlf * 8 + v;
        int col = bn + wn * 32 + j * 16 + l16;
        if constexpr (OUT_F32)
          ((float*)C)[(size_t)row * N + col] = acc[i][j][v];
        else
          ((unsigned short*)C)[(size_t)row * N + col] = f2bf(acc[i][j][v]);
      }
}

// ---------------------------------------------------------------------------
// Flash attention (causal, GQA 4:1). One wave handles 16 query rows.
// Block = 128 threads = 4 independent waves. Grid (S/64, H, B).
// Q,K,V bf16; online softmax in fp32; output bf16 [b*S+q, h*64+hd].
// ---------------------------------------------------------------------------
__global__ __launch_bounds__(128)
void k_flash_attn(const unsigned short* __restrict__ qb,
                  const unsigned short* __restrict__ kb,
                  const unsigned short* __restrict__ vb,
                  unsigned short* __restrict__ ob) {
  constexpr int S = 1024, HD = 64, QSTR = 32 * 64, KSTR = 8 * 64, NREP = 4;
  __shared__ unsigned short Pbuf[4][16][32 + 8];   // per-wave P tile [q][key]
  __shared__ unsigned short Vs[4][64][32 + 8];     // per-wave V^T [hd][key]

  const int wave = threadIdx.x >> 5;
  const int lane = threadIdx.x & 31;
  const int hlf  = lane >> 4;
  const int l16  = lane & 15;
  const int b    = blockIdx.z;
  const int h    = blockIdx.y;
  const int q0   = blockIdx.x * 64 + wave * 16;
  const int kvh  = h / NREP;

  // ---- load Q fragments (16 rows x 64 hd = 2 A-frags), pre-scaled 1/sqrt(64)
  v16bf qa[2];
  {
    const unsigned short* qp = qb + (size_t)(b * S + q0 + l16) * QSTR + h * HD;
#pragma unroll
    for (int s = 0; s < 2; ++s)
#pragma unroll
      for (int e = 0; e < 8; ++e) {
        qa[s][e]     = us2bf(f2bf(bf2f(qp[s * 32 + hlf * 8 + e]) * 0.125f));
        qa[s][8 + e] = us2bf(f2bf(bf2f(qp[s * 32 + 16 + hlf * 8 + e]) * 0.125f));
      }
  }

  const v8f vzero = {0.f, 0.f, 0.f, 0.f, 0.f, 0.f, 0.f, 0.f};
  v8f o[4];
#pragma unroll
  for (int j = 0; j < 4; ++j) o[j] = vzero;
  float mrow[8], lrow[8];
#pragma unroll
  for (int v = 0; v < 8; ++v) { mrow[v] = -__builtin_inff(); lrow[v] = 0.f; }

  for (int kc = 0; kc < q0 + 16; kc += 32) {
    // ---- stage V chunk transposed: lane owns key kc+lane, writes Vs[hd][key]
    {
      const unsigned short* vp =
          vb + (size_t)(b * S + kc + lane) * KSTR + kvh * HD;
#pragma unroll
      for (int d = 0; d < 64; ++d) Vs[wave][d][lane] = vp[d];
    }

    // ---- scores: two 16x16 tiles (32 keys), contraction over hd=64 (2 steps)
    v8f sc[2];
#pragma unroll
    for (int t = 0; t < 2; ++t) {
      const unsigned short* kp =
          kb + (size_t)(b * S + kc + t * 16 + l16) * KSTR + kvh * HD;
      v16bf kf0, kf1;
#pragma unroll
      for (int e = 0; e < 16; ++e) {
        kf0[e] = us2bf(kp[hlf * 8 + (e & 7) + ((e >> 3) << 4)]);        // hd 0..31
        kf1[e] = us2bf(kp[32 + hlf * 8 + (e & 7) + ((e >> 3) << 4)]);   // hd 32..63
      }
      v8f s = vzero;
      s = __builtin_amdgcn_wmma_f32_16x16x32_bf16(false, qa[0], false, kf0,
                                                  (short)0, s, false, false);
      s = __builtin_amdgcn_wmma_f32_16x16x32_bf16(false, qa[1], false, kf1,
                                                  (short)0, s, false, false);
      sc[t] = s;
    }

    // ---- causal mask + online softmax (rows live in 16-lane halves)
#pragma unroll
    for (int v = 0; v < 8; ++v) {
      int qrow = q0 + hlf * 8 + v;
      float s0 = sc[0][v]; if (kc + l16 > qrow)      s0 = -__builtin_inff();
      float s1 = sc[1][v]; if (kc + 16 + l16 > qrow) s1 = -__builtin_inff();
      float mx = fmaxf(s0, s1);
#pragma unroll
      for (int off = 1; off < 16; off <<= 1)
        mx = fmaxf(mx, __shfl_xor(mx, off, 32));
      float mnew = fmaxf(mrow[v], mx);
      float sf   = __expf(mrow[v] - mnew);
      mrow[v] = mnew;
      float p0 = __expf(s0 - mnew);
      float p1 = __expf(s1 - mnew);
      float rs = p0 + p1;
#pragma unroll
      for (int off = 1; off < 16; off <<= 1) rs += __shfl_xor(rs, off, 32);
      lrow[v] = lrow[v] * sf + rs;
#pragma unroll
      for (int j = 0; j < 4; ++j) o[j][v] = o[j][v] * sf;
      Pbuf[wave][hlf * 8 + v][l16]      = f2bf(p0);
      Pbuf[wave][hlf * 8 + v][16 + l16] = f2bf(p1);
    }

    // LDS within a wave is in-order; wait anyway before the re-layout reads.
    asm volatile("s_wait_dscnt 0" ::: "memory");

    // ---- P (16x32) A-fragment from LDS
    v16bf pa;
    {
      const unsigned short* pp = &Pbuf[wave][l16][hlf * 8];
#pragma unroll
      for (int e = 0; e < 8; ++e) {
        pa[e]     = us2bf(pp[e]);        // key = hlf*8 + e
        pa[8 + e] = us2bf(pp[16 + e]);   // key = 16 + hlf*8 + e
      }
    }

    // ---- O += P @ V  (4 hd tiles of 16)
#pragma unroll
    for (int j = 0; j < 4; ++j) {
      const unsigned short* vp2 = &Vs[wave][j * 16 + l16][hlf * 8];
      v16bf vf;
#pragma unroll
      for (int e = 0; e < 16; ++e) vf[e] = us2bf(vp2[(e & 7) + ((e >> 3) << 4)]);  // key = hlf*8 + e (e<8), 16 + hlf*8 + e-8
      o[j] = __builtin_amdgcn_wmma_f32_16x16x32_bf16(false, pa, false, vf,
                                                     (short)0, o[j], false, false);
    }
  }

  // ---- normalize and store bf16
#pragma unroll
  for (int v = 0; v < 8; ++v) {
    float inv = 1.0f / lrow[v];
    unsigned short* op =
        ob + (size_t)(b * S + q0 + hlf * 8 + v) * QSTR + h * HD;
#pragma unroll
    for (int j = 0; j < 4; ++j) op[j * 16 + l16] = f2bf(o[j][v] * inv);
  }
}

// ---------------------------------------------------------------------------
// Launcher
// ---------------------------------------------------------------------------
extern "C" void kernel_launch(void* const* d_in, const int* in_sizes, int n_in,
                              void* d_out, int out_size, void* d_ws, size_t ws_size,
                              hipStream_t stream) {
  (void)in_sizes; (void)n_in; (void)out_size; (void)ws_size;
  constexpr int Bsz = 4, S = 1024, D = 2048, H = 32, KH = 8, HD = 64;
  constexpr int M = Bsz * S;                 // 4096
  const float* x    = (const float*)d_in[0];
  const float* wq   = (const float*)d_in[1];
  const float* wk   = (const float*)d_in[2];
  const float* wv   = (const float*)d_in[3];
  const float* wo   = (const float*)d_in[4];
  const float* fcos = (const float*)d_in[5];
  const float* fsin = (const float*)d_in[6];
  float* out        = (float*)d_out;

  // workspace carve-up (bf16 elements)
  unsigned short* xb  = (unsigned short*)d_ws;          // 4096x2048
  unsigned short* wqb = xb  + (size_t)M * D;            // 2048x2048
  unsigned short* wkb = wqb + (size_t)D * (H * HD);     // 2048x512
  unsigned short* wvb = wkb + (size_t)D * (KH * HD);    // 2048x512
  unsigned short* wob = wvb + (size_t)D * (KH * HD);    // 2048x2048
  unsigned short* qb  = wob + (size_t)(H * HD) * D;     // 4096x2048
  unsigned short* kbf = qb  + (size_t)M * (H * HD);     // 4096x512
  unsigned short* vbf = kbf + (size_t)M * (KH * HD);    // 4096x512
  unsigned short* ab  = vbf + (size_t)M * (KH * HD);    // 4096x2048

  auto cvt = [&](const float* src, unsigned short* dst, int n) {
    k_f32_to_bf16<<<(n + 255) / 256, 256, 0, stream>>>(src, dst, n);
  };
  cvt(x,  xb,  M * D);
  cvt(wq, wqb, D * H * HD);
  cvt(wk, wkb, D * KH * HD);
  cvt(wv, wvb, D * KH * HD);
  cvt(wo, wob, H * HD * D);

  // Q/K/V projections (bf16 out)
  k_gemm_bf16<false><<<dim3(M / 128, (H * HD) / 64), 256, 0, stream>>>(
      xb, wqb, qb, M, H * HD, D);
  k_gemm_bf16<false><<<dim3(M / 128, (KH * HD) / 64), 256, 0, stream>>>(
      xb, wkb, kbf, M, KH * HD, D);
  k_gemm_bf16<false><<<dim3(M / 128, (KH * HD) / 64), 256, 0, stream>>>(
      xb, wvb, vbf, M, KH * HD, D);

  // RoPE on Q and K
  k_rope_bf16<<<(M * (H * HD / 2) + 255) / 256, 256, 0, stream>>>(
      qb, fcos, fsin, H, M * (H * HD / 2));
  k_rope_bf16<<<(M * (KH * HD / 2) + 255) / 256, 256, 0, stream>>>(
      kbf, fcos, fsin, KH, M * (KH * HD / 2));

  // Flash attention
  k_flash_attn<<<dim3(S / 64, H, Bsz), 128, 0, stream>>>(qb, kbf, vbf, ab);

  // Output projection (fp32 out)
  k_gemm_bf16<true><<<dim3(M / 128, D / 64), 256, 0, stream>>>(
      ab, wob, out, M, D, H * HD);
}
